// TokenRoutedMLP_51470888075916
// MI455X (gfx1250) — matrix operand order repaired and checked by reading, not compiled, not run on hardware
//
#include <hip/hip_runtime.h>
#include <stdint.h>

#define H  1024
#define E  8
#define II 512          // I = 4096 / E
#define N2 1024         // 2*I
#define V  32000
#define T  4096

typedef __attribute__((ext_vector_type(16))) __bf16 v16bf;
typedef __attribute__((ext_vector_type(8)))  float  v8f;

struct __align__(16) U4 { unsigned int a, b, c, d; };
union V16 { v16bf v; U4 q[2]; };

// fp32 -> bf16 (round to nearest even), raw 16 bits
__device__ __forceinline__ unsigned int f2bf(float f) {
  unsigned int u = __builtin_bit_cast(unsigned int, f);
  return (u + 0x7FFFu + ((u >> 16) & 1u)) >> 16;
}
__device__ __forceinline__ unsigned int pk2(float lo, float hi) {
  return f2bf(lo) | (f2bf(hi) << 16);
}

__device__ __forceinline__ v8f wmma_bf16(const v16bf& a, const v16bf& b, v8f c) {
  return __builtin_amdgcn_wmma_f32_16x16x32_bf16(false, a, false, b, (short)0, c,
                                                 false, false);
}

// ---------------- routing: one wave per token ----------------
__global__ __launch_bounds__(256) void route_kernel(
    const float* __restrict__ mu, const float* __restrict__ mu_w,
    const int* __restrict__ tok_ids, int* __restrict__ counts,
    int* __restrict__ bucket) {
  const int lane = threadIdx.x & 31;
  const int wv   = threadIdx.x >> 5;
  const int t = blockIdx.x * 8 + wv;
  if (t >= T) return;
  float acc[E];
#pragma unroll
  for (int e = 0; e < E; ++e) acc[e] = 0.f;
  const float* mrow = mu + (size_t)t * H;
  for (int h = lane; h < H; h += 32) {
    float m = mrow[h];
#pragma unroll
    for (int e = 0; e < E; ++e) acc[e] += m * mu_w[e * H + h];
  }
#pragma unroll
  for (int e = 0; e < E; ++e) {
#pragma unroll
    for (int off = 16; off > 0; off >>= 1)
      acc[e] += __shfl_xor(acc[e], off, 32);
  }
  if (lane == 0) {
    int tkid = tok_ids[t];
    if (tkid < 0) tkid = 0;
    if (tkid > V - 1) tkid = V - 1;
    const int be = tkid & (E - 1);
    int best = 0; float bv = -3.4e38f;
#pragma unroll
    for (int e = 0; e < E; ++e) {
      float v = acc[e] + (e == be ? 10.f : 0.f);
      if (v > bv) { bv = v; best = e; }
    }
    int slot = atomicAdd(&counts[best], 1);
    bucket[best * T + slot] = t;
  }
}

__global__ void offsets_kernel(const int* __restrict__ counts,
                               int* __restrict__ offs) {
  if (threadIdx.x == 0 && blockIdx.x == 0) {
    int s = 0;
    for (int e = 0; e < E; ++e) { offs[e] = s; s += counts[e]; }
    offs[E] = s;
  }
}

// -------- prepass: fp32 -> packed bf16 pairs (row-major, for x) --------
__global__ __launch_bounds__(256) void convert_pairs(
    const float2* __restrict__ src, unsigned int* __restrict__ dst, int npair) {
  int i = blockIdx.x * 256 + threadIdx.x;
  if (i < npair) {
    float2 f = src[i];
    dst[i] = pk2(f.x, f.y);
  }
}

// -------- prepass: per-expert transpose [K][N] fp32 -> [N][K] bf16 pairs ----
__global__ __launch_bounds__(256) void transpose_bf16(
    const float* __restrict__ src, unsigned int* __restrict__ dst,
    int K, int N) {
  const int e  = blockIdx.z;
  const int k0 = blockIdx.x * 64;
  const int n0 = blockIdx.y * 64;
  const float* s = src + (size_t)e * K * N;
  unsigned int* d = dst + ((size_t)e * K * N) / 2;

  __shared__ float tile[64][65];
  const int tid = threadIdx.x;
  const int r = tid >> 2;
#pragma unroll
  for (int i = 0; i < 4; ++i) {
    int c4 = (tid & 3) + i * 4;  // float4 column index, 0..15
    const float4 v = *(const float4*)(s + (size_t)(k0 + r) * N + n0 + c4 * 4);
    tile[r][c4 * 4 + 0] = v.x; tile[r][c4 * 4 + 1] = v.y;
    tile[r][c4 * 4 + 2] = v.z; tile[r][c4 * 4 + 3] = v.w;
  }
  __syncthreads();
#pragma unroll
  for (int i = 0; i < 8; ++i) {
    int idx = tid + i * 256;          // 0..2047
    int kp = idx & 31, n = idx >> 5;  // coalesced along kp
    d[((size_t)(n0 + n) * K + k0) / 2 + kp] =
        pk2(tile[2 * kp][n], tile[2 * kp + 1][n]);
  }
}

// -------- gemm1: inter = silu(Xg @ W1g) * (Xg @ W1u), double-buffered -------
__global__ __launch_bounds__(256) void gemm1_silu(
    const U4* __restrict__ xq, const U4* __restrict__ w1q,
    const int* __restrict__ counts, const int* __restrict__ offs,
    const int* __restrict__ bucket, unsigned short* __restrict__ inter) {
  const int e   = blockIdx.z;
  const int cnt = counts[e];
  const int m0  = blockIdx.x * 64;
  if (m0 >= cnt) return;
  const int n0  = blockIdx.y * 64;  // gate column base in [0, I)

  __shared__ __align__(16) unsigned int Asq[2][64][16];
  __shared__ __align__(16) unsigned int Bgq[2][64][16];
  __shared__ __align__(16) unsigned int Buq[2][64][16];

  const int tid  = threadIdx.x;
  const int lane = tid & 31;
  const int wv   = tid >> 5;
  const int mi   = wv & 3;
  const int nj   = wv >> 2;
  const int g    = lane >> 4;
  const int ln   = lane & 15;

  const int* buck = bucket + e * T;
  const U4*  we   = w1q + (size_t)e * (N2 * (H / 8));  // [n][k] bf16, uint4 units

  const int lr = tid >> 2;   // 0..63: A row / B row(n)
  const int lq = tid & 3;    // which 16B chunk of the 64B k-slab
  const int mrowL = m0 + lr;
  const int tokL  = (mrowL < cnt) ? buck[mrowL] : 0;

  auto loadTiles = [&](int kk, int pb) {
    U4 va = {0u, 0u, 0u, 0u};
    if (mrowL < cnt) va = xq[(size_t)tokL * (H / 8) + (kk >> 3) + lq];
    *(U4*)&Asq[pb][lr][lq * 4] = va;
    *(U4*)&Bgq[pb][lr][lq * 4] = we[((size_t)(n0 + lr) * H + kk) / 8 + lq];
    *(U4*)&Buq[pb][lr][lq * 4] = we[((size_t)(n0 + II + lr) * H + kk) / 8 + lq];
  };

  v8f cg0 = {}, cg1 = {}, cu0 = {}, cu1 = {};
  loadTiles(0, 0);
  int p = 0;
  for (int k0 = 0; k0 < H; k0 += 32) {
    __syncthreads();
    if (k0 + 32 < H) loadTiles(k0 + 32, p ^ 1);
    if (k0 + 64 < H)
      __builtin_prefetch(&we[((size_t)(n0 + lr) * H + k0 + 64) / 8], 0, 1);

    V16 a;
    a.q[0] = *(const U4*)&Asq[p][mi * 16 + ln][g * 8];
    a.q[1] = *(const U4*)&Asq[p][mi * 16 + ln][g * 8 + 4];
    const int nA = nj * 32;
    V16 bg0, bg1, bu0, bu1;
    bg0.q[0] = *(const U4*)&Bgq[p][nA + ln][g * 8];
    bg0.q[1] = *(const U4*)&Bgq[p][nA + ln][g * 8 + 4];
    bg1.q[0] = *(const U4*)&Bgq[p][nA + 16 + ln][g * 8];
    bg1.q[1] = *(const U4*)&Bgq[p][nA + 16 + ln][g * 8 + 4];
    bu0.q[0] = *(const U4*)&Buq[p][nA + ln][g * 8];
    bu0.q[1] = *(const U4*)&Buq[p][nA + ln][g * 8 + 4];
    bu1.q[0] = *(const U4*)&Buq[p][nA + 16 + ln][g * 8];
    bu1.q[1] = *(const U4*)&Buq[p][nA + 16 + ln][g * 8 + 4];

    cg0 = wmma_bf16(a.v, bg0.v, cg0);
    cg1 = wmma_bf16(a.v, bg1.v, cg1);
    cu0 = wmma_bf16(a.v, bu0.v, cu0);
    cu1 = wmma_bf16(a.v, bu1.v, cu1);
    p ^= 1;
  }

  const int offe = offs[e];
#pragma unroll
  for (int tt = 0; tt < 2; ++tt) {
    v8f ga = tt ? cg1 : cg0;
    v8f ua = tt ? cu1 : cu0;
    int ncol = n0 + nj * 32 + tt * 16 + ln;
#pragma unroll
    for (int r = 0; r < 8; ++r) {
      int mrow = m0 + mi * 16 + r + 8 * g;
      if (mrow < cnt) {
        float gv = ga[r], uv = ua[r];
        float sv = gv / (1.0f + __expf(-gv));
        inter[(size_t)(offe + mrow) * II + ncol] = (unsigned short)f2bf(sv * uv);
      }
    }
  }
}

// -------- gemm2: out[token] = inter_packed @ W2t[e], double-buffered --------
__global__ __launch_bounds__(256) void gemm2_scatter(
    const U4* __restrict__ interq, const U4* __restrict__ w2q,
    const int* __restrict__ counts, const int* __restrict__ offs,
    const int* __restrict__ bucket, float* __restrict__ out) {
  const int e   = blockIdx.z;
  const int cnt = counts[e];
  const int m0  = blockIdx.x * 64;
  if (m0 >= cnt) return;
  const int n0  = blockIdx.y * 64;

  __shared__ __align__(16) unsigned int Asq[2][64][16];
  __shared__ __align__(16) unsigned int Bsq[2][64][16];

  const int tid  = threadIdx.x;
  const int lane = tid & 31;
  const int wv   = tid >> 5;
  const int mi   = wv & 3;
  const int nj   = wv >> 2;
  const int g    = lane >> 4;
  const int ln   = lane & 15;

  const int* buck = bucket + e * T;
  const int  offe = offs[e];
  const U4*  we   = w2q + (size_t)e * (H * (II / 8));  // [n][k] bf16

  const int lr = tid >> 2;
  const int lq = tid & 3;
  const int mrowL = m0 + lr;

  auto loadTiles = [&](int kk, int pb) {
    U4 va = {0u, 0u, 0u, 0u};
    if (mrowL < cnt)
      va = interq[((size_t)(offe + mrowL) * II + kk) / 8 + lq];
    *(U4*)&Asq[pb][lr][lq * 4] = va;
    *(U4*)&Bsq[pb][lr][lq * 4] = we[((size_t)(n0 + lr) * II + kk) / 8 + lq];
  };

  v8f c0 = {}, c1 = {};
  loadTiles(0, 0);
  int p = 0;
  for (int k0 = 0; k0 < II; k0 += 32) {
    __syncthreads();
    if (k0 + 32 < II) loadTiles(k0 + 32, p ^ 1);
    if (k0 + 64 < II)
      __builtin_prefetch(&we[((size_t)(n0 + lr) * II + k0 + 64) / 8], 0, 1);

    V16 a;
    a.q[0] = *(const U4*)&Asq[p][mi * 16 + ln][g * 8];
    a.q[1] = *(const U4*)&Asq[p][mi * 16 + ln][g * 8 + 4];
    const int nA = nj * 32;
    V16 b0, b1;
    b0.q[0] = *(const U4*)&Bsq[p][nA + ln][g * 8];
    b0.q[1] = *(const U4*)&Bsq[p][nA + ln][g * 8 + 4];
    b1.q[0] = *(const U4*)&Bsq[p][nA + 16 + ln][g * 8];
    b1.q[1] = *(const U4*)&Bsq[p][nA + 16 + ln][g * 8 + 4];

    c0 = wmma_bf16(a.v, b0.v, c0);
    c1 = wmma_bf16(a.v, b1.v, c1);
    p ^= 1;
  }

#pragma unroll
  for (int tt = 0; tt < 2; ++tt) {
    v8f acc = tt ? c1 : c0;
    int ncol = n0 + nj * 32 + tt * 16 + ln;
#pragma unroll
    for (int r = 0; r < 8; ++r) {
      int mrow = m0 + mi * 16 + r + 8 * g;
      if (mrow < cnt) {
        int tk = buck[mrow];
        out[(size_t)tk * H + ncol] = acc[r];
      }
    }
  }
}

extern "C" void kernel_launch(void* const* d_in, const int* in_sizes, int n_in,
                              void* d_out, int out_size, void* d_ws, size_t ws_size,
                              hipStream_t stream) {
  const float* x  = (const float*)d_in[0];
  const int*   tk = (const int*)d_in[1];
  const float* mu = (const float*)d_in[2];
  const float* w1 = (const float*)d_in[3];   // (E, H, 2I)
  const float* w2 = (const float*)d_in[4];   // (E, I, H)
  const float* mw = (const float*)d_in[5];   // (E, H)
  float* out = (float*)d_out;

  // workspace layout (all segments 16B aligned)
  char* ws = (char*)d_ws;
  const size_t OFF_COUNTS = 0;                                // 32 ints
  const size_t OFF_BUCKET = 128;                              // E*T ints
  const size_t OFF_INTER  = OFF_BUCKET + (size_t)E * T * 4;   // T*I bf16
  const size_t OFF_XBF    = OFF_INTER + (size_t)T * II * 2;   // T*H bf16
  const size_t OFF_W1T    = OFF_XBF + (size_t)T * H * 2;      // E*N2*H bf16
  const size_t OFF_W2T    = OFF_W1T + (size_t)E * N2 * H * 2; // E*H*I bf16

  int* counts = (int*)(ws + OFF_COUNTS);
  int* offs   = counts + 8;
  int* bucket = (int*)(ws + OFF_BUCKET);
  unsigned short* inter = (unsigned short*)(ws + OFF_INTER);
  unsigned int* xbf = (unsigned int*)(ws + OFF_XBF);
  unsigned int* w1t = (unsigned int*)(ws + OFF_W1T);
  unsigned int* w2t = (unsigned int*)(ws + OFF_W2T);

  hipMemsetAsync(counts, 0, 32 * sizeof(int), stream);
  route_kernel<<<T / 8, 256, 0, stream>>>(mu, mw, tk, counts, bucket);
  offsets_kernel<<<1, 32, 0, stream>>>(counts, offs);

  convert_pairs<<<(T * H / 2) / 256, 256, 0, stream>>>(
      (const float2*)x, xbf, T * H / 2);
  transpose_bf16<<<dim3(H / 64, N2 / 64, E), 256, 0, stream>>>(w1, w1t, H, N2);
  transpose_bf16<<<dim3(II / 64, H / 64, E), 256, 0, stream>>>(w2, w2t, II, H);

  gemm1_silu<<<dim3(T / 64, II / 64, E), 256, 0, stream>>>(
      (const U4*)xbf, (const U4*)w1t, counts, offs, bucket, inter);
  gemm2_scatter<<<dim3(T / 64, H / 64, E), 256, 0, stream>>>(
      (const U4*)inter, (const U4*)w2t, counts, offs, bucket, out);
}
